// CustomModel_4372276707887
// MI455X (gfx1250) — matrix operand order repaired and checked by reading, not compiled, not run on hardware
//
#include <hip/hip_runtime.h>
#include <hip/hip_bf16.h>
#include <stdint.h>

// ---------------------------------------------------------------------------
// MI455X (gfx1250) fused GCLSTM-block forward.
//
// Dead-code analysis of the reference (exact, not approximate):
//   * H=C=0  =>  all ChebConv terms are 0 (+zero bias), edge data unused
//   * F-gate multiplied by C=0  => unused
//   * L == R (same params/input), second block round discarded
// => per-node fused MLP; one wave handles 16 nodes, all GEMMs via
//    v_wmma_f32_16x16x32_f16 with f16 weights resident in LDS.
//   Gate nonlinearities use CDNA5's native V_TANH_F32 transcendental
//   (sigma(z) = 0.5 + 0.5*tanh(z/2)), which co-executes with the WMMA pipe.
// ---------------------------------------------------------------------------

typedef __attribute__((ext_vector_type(16))) _Float16 v16h;
typedef __attribute__((ext_vector_type(8)))  float    v8f;

#define N_NODES 50000
#define F_IN    64
#define TILES   ((N_NODES + 15) / 16)       // 3125 (exact: 3125*16 == 50000)
#define WAVES_PER_BLOCK 4
#define BLOCK_THREADS   (WAVES_PER_BLOCK * 32)

#define STG 70   // per-wave staging row stride in halves (bank-conflict free)
#define KS1 70   // column stride (halves) for K=64 weight panels
#define KS2 38   // column stride (halves) for K<=32 weight panels

struct Params {
  const float *Wi1,*Wc1,*Wo1,*bi1,*bc1,*bo1,*cbi1,*cbc1,*cbo1,*wco1;
  const float *c1w,*c1b;
  const float *Wi2,*Wc2,*Wo2,*bi2,*bc2,*bo2,*cbi2,*cbc2,*cbo2,*wco2;
  const float *c2w,*c2b,*linw,*linb,*x;
};

struct __align__(16) Smem {
  _Float16 wi1[64*KS1], wc1[64*KS1], wo1[64*KS1];   // g1 gates, col-major f16
  _Float16 c1 [32*KS1];                             // c1_w col-major
  _Float16 wi2[16*KS2], wc2[16*KS2], wo2[16*KS2];   // g2 gates (K=32)
  _Float16 c2 [64*KS2];                             // c2_w, K padded 16->32
  _Float16 stgA[WAVES_PER_BLOCK][16*STG];           // per-wave activation tiles
  _Float16 stgB[WAVES_PER_BLOCK][16*STG];
};

__device__ __forceinline__ void lds_waveorder() {
  // DS ops from one wave are processed in order; this is a compiler barrier
  // (prevents IR reordering of f16 stores vs u32 reloads) + explicit wait.
  asm volatile("s_wait_dscnt 0" ::: "memory");
}

// CDNA5 native transcendental tanh (TRANS32).
__device__ __forceinline__ float dev_tanh(float z) {
#if __has_builtin(__builtin_amdgcn_tanhf)
  return __builtin_amdgcn_tanhf(z);
#elif __has_builtin(__builtin_amdgcn_tanh_f32)
  return __builtin_amdgcn_tanh_f32(z);
#else
  float r;
  // ISA 7.4: TRANS result needs 1 independent op / V_NOP before use;
  // the compiler cannot see through asm, so pad explicitly.
  asm volatile("v_tanh_f32 %0, %1\n\tv_nop\n\tv_nop" : "=v"(r) : "v"(z));
  return r;
#endif
}
__device__ __forceinline__ float dev_sigmoid(float z) {
  return fmaf(0.5f, dev_tanh(0.5f * z), 0.5f);   // exact identity
}

// A fragment (16x32 f16, ISA 7.12.2): lane = {g=lane>>4, m=lane&15},
// VGPR v holds K = kt*32 + g*8 + (v<4 ? 2v : 16+2(v-4)) .. +1
__device__ __forceinline__ v16h load_a(const _Float16* stg, int m, int g, int kt) {
  union { v16h h; uint32_t u[8]; } a;
  const _Float16* row = stg + m * STG + kt * 32 + g * 8;
#pragma unroll
  for (int v = 0; v < 8; ++v) {
    int k = (v < 4) ? (2 * v) : (16 + 2 * (v - 4));
    a.u[v] = *(const uint32_t*)(row + k);
  }
  return a.h;
}

// B fragment (32x16 f16): same K order as A: VGPR v holds
// K = kt*32 + g*8 + (v<4 ? 2v : 16+2(v-4)) .. +1; column n = lane&15. Weights are column-major.
__device__ __forceinline__ v16h load_b(const _Float16* wcm, int ks, int col, int g, int kt) {
  union { v16h h; uint32_t u[8]; } b;
  const _Float16* p = wcm + col * ks + kt * 32 + g * 8;
#pragma unroll
  for (int v = 0; v < 8; ++v) b.u[v] = *(const uint32_t*)(p + ((v < 4) ? (2 * v) : (16 + 2 * (v - 4))));
  return b.h;
}

__device__ __forceinline__ v8f wmma16(v16h a, v16h b, v8f c) {
  return __builtin_amdgcn_wmma_f32_16x16x32_f16(false, a, false, b, (short)0, c,
                                                false, false);
}

__device__ __forceinline__ void stage_cm(const float* W, _Float16* dst,
                                         int K, int N, int ks) {
  for (int idx = threadIdx.x; idx < K * N; idx += blockDim.x) {
    int k = idx / N, c = idx - k * N;               // W is row-major [K][N]
    dst[c * ks + k] = (_Float16)W[idx];
  }
}

__global__ __launch_bounds__(BLOCK_THREADS)
void gclstm_fused_kernel(Params p, float* __restrict__ out) {
  __shared__ Smem s;

  stage_cm(p.Wi1, s.wi1, 64, 64, KS1);
  stage_cm(p.Wc1, s.wc1, 64, 64, KS1);
  stage_cm(p.Wo1, s.wo1, 64, 64, KS1);
  stage_cm(p.c1w, s.c1,  64, 32, KS1);
  stage_cm(p.Wi2, s.wi2, 32, 16, KS2);
  stage_cm(p.Wc2, s.wc2, 32, 16, KS2);
  stage_cm(p.Wo2, s.wo2, 32, 16, KS2);
  stage_cm(p.c2w, s.c2,  16, 64, KS2);
  // zero-pad c2 K rows 16..31 (WMMA K is 32)
  for (int idx = threadIdx.x; idx < 64 * 16; idx += blockDim.x) {
    int c = idx >> 4, k = 16 + (idx & 15);
    s.c2[c * KS2 + k] = (_Float16)0.f;
  }
  __syncthreads();

  const int lane = threadIdx.x & 31;
  const int wave = threadIdx.x >> 5;
  const int g    = lane >> 4;     // half-wave group
  const int l15  = lane & 15;     // A row / D column within tile
  const int tile = blockIdx.x * WAVES_PER_BLOCK + wave;   // wave-uniform
  if (tile >= TILES) return;      // uniform exit; EXEC stays all-ones below
  const int base = tile * 16;

  _Float16* sA = s.stgA[wave];
  _Float16* sB = s.stgB[wave];

  // ---- stage x tile (16 contiguous rows of 64 f32) -> f16 LDS ------------
  const float4* xv = (const float4*)(p.x + (size_t)base * F_IN);
#pragma unroll
  for (int it = 0; it < 8; ++it) {
    int idx = it * 32 + lane;                 // 256 float4 = 1024 floats
    float4 v = xv[idx];
    int r = idx >> 4, c = (idx & 15) << 2;
    _Float16* q = sA + r * STG + c;
    q[0] = (_Float16)v.x; q[1] = (_Float16)v.y;
    q[2] = (_Float16)v.z; q[3] = (_Float16)v.w;
  }
  lds_waveorder();

  // ---- GCLSTM layer 1: 64 -> 64, fused relu ------------------------------
  {
    v16h A0 = load_a(sA, l15, g, 0);
    v16h A1 = load_a(sA, l15, g, 1);
#pragma unroll
    for (int ct = 0; ct < 4; ++ct) {
      v8f aI = {}, aT = {}, aO = {};
      int col = ct * 16 + l15;
      v16h b;
      b = load_b(s.wi1, KS1, col, g, 0); aI = wmma16(A0, b, aI);
      b = load_b(s.wi1, KS1, col, g, 1); aI = wmma16(A1, b, aI);
      b = load_b(s.wc1, KS1, col, g, 0); aT = wmma16(A0, b, aT);
      b = load_b(s.wc1, KS1, col, g, 1); aT = wmma16(A1, b, aT);
      b = load_b(s.wo1, KS1, col, g, 0); aO = wmma16(A0, b, aO);
      b = load_b(s.wo1, KS1, col, g, 1); aO = wmma16(A1, b, aO);
      float bi  = p.bi1[col] + p.cbi1[col];
      float bc  = p.bc1[col] + p.cbc1[col];
      float bo  = p.bo1[col] + p.cbo1[col];
      float wco = p.wco1[col];
#pragma unroll
      for (int v = 0; v < 8; ++v) {
        float I = dev_sigmoid(aI[v] + bi);
        float T = dev_tanh(aT[v] + bc);
        float C = I * T;
        float O = dev_sigmoid(aO[v] + bo + wco * C);
        float H = fmaxf(O * dev_tanh(C), 0.f);
        sB[(v + 8 * g) * STG + col] = (_Float16)H;
      }
    }
  }
  lds_waveorder();

  // ---- c1: 64 -> 32, bias + relu -----------------------------------------
  {
    v16h A0 = load_a(sB, l15, g, 0);
    v16h A1 = load_a(sB, l15, g, 1);
#pragma unroll
    for (int ct = 0; ct < 2; ++ct) {
      v8f acc = {};
      int col = ct * 16 + l15;
      v16h b;
      b = load_b(s.c1, KS1, col, g, 0); acc = wmma16(A0, b, acc);
      b = load_b(s.c1, KS1, col, g, 1); acc = wmma16(A1, b, acc);
      float cb = p.c1b[col];
#pragma unroll
      for (int v = 0; v < 8; ++v) {
        float h = fmaxf(acc[v] + cb, 0.f);
        sA[(v + 8 * g) * STG + col] = (_Float16)h;   // cols 0..31
      }
    }
  }
  lds_waveorder();

  // ---- GCLSTM layer 2: 32 -> 16, fused relu; zero-pad cols 16..31 --------
  {
    v16h A0 = load_a(sA, l15, g, 0);   // reads K=0..31 only
    v8f aI = {}, aT = {}, aO = {};
    v16h b;
    b = load_b(s.wi2, KS2, l15, g, 0); aI = wmma16(A0, b, aI);
    b = load_b(s.wc2, KS2, l15, g, 0); aT = wmma16(A0, b, aT);
    b = load_b(s.wo2, KS2, l15, g, 0); aO = wmma16(A0, b, aO);
    int col = l15;
    float bi  = p.bi2[col] + p.cbi2[col];
    float bc  = p.bc2[col] + p.cbc2[col];
    float bo  = p.bo2[col] + p.cbo2[col];
    float wco = p.wco2[col];
#pragma unroll
    for (int v = 0; v < 8; ++v) {
      float I = dev_sigmoid(aI[v] + bi);
      float T = dev_tanh(aT[v] + bc);
      float C = I * T;
      float O = dev_sigmoid(aO[v] + bo + wco * C);
      float H = fmaxf(O * dev_tanh(C), 0.f);
      sB[(v + 8 * g) * STG + col]      = (_Float16)H;
      sB[(v + 8 * g) * STG + col + 16] = (_Float16)0.f;  // K-pad for c2
    }
  }
  lds_waveorder();

  // ---- c2: 16 -> 64 (+bias, relu) fused with lin: 64 -> 1 ----------------
  {
    v16h A0 = load_a(sB, l15, g, 0);
    float r[8] = {0.f, 0.f, 0.f, 0.f, 0.f, 0.f, 0.f, 0.f};
#pragma unroll
    for (int ct = 0; ct < 4; ++ct) {
      v8f acc = {};
      v16h b = load_b(s.c2, KS2, ct * 16 + l15, g, 0);
      acc = wmma16(A0, b, acc);
      int col = ct * 16 + l15;
      float cb = p.c2b[col], lw = p.linw[col];
#pragma unroll
      for (int v = 0; v < 8; ++v)
        r[v] += fmaxf(acc[v] + cb, 0.f) * lw;
    }
    // reduce across the 16 lanes of each half-group (rows v+8g)
#pragma unroll
    for (int off = 8; off >= 1; off >>= 1)
#pragma unroll
      for (int v = 0; v < 8; ++v) r[v] += __shfl_xor(r[v], off, 32);
    if (l15 == 0) {
      float lb = p.linb[0];
#pragma unroll
      for (int v = 0; v < 8; ++v) {
        int node = base + v + 8 * g;
        float o = r[v] + lb;
        out[node]           = o;   // L half of concat
        out[N_NODES + node] = o;   // R half (L == R)
      }
    }
  }
}

extern "C" void kernel_launch(void* const* d_in, const int* in_sizes, int n_in,
                              void* d_out, int out_size, void* d_ws, size_t ws_size,
                              hipStream_t stream) {
  (void)out_size; (void)d_ws; (void)ws_size;
  auto F = [&](int i) { return (const float*)d_in[i]; };
  Params p;
  // Detect flatten order (pure function of inputs):
  //  - jax tree_leaves (sorted dict keys):   in[0] = edge_index (2*800000)
  //  - recursive insertion order:            in[0] = g1.W_i     (64*64)
  bool insertion = (n_in > 0 && in_sizes[0] == 64 * 64);
  if (!insertion) {
    // alphabetical: edge_index, edge_weight, c1_b, c1_w, c2_b, c2_w,
    // g1{W_c,W_f,W_i,W_o, b_c..b_o, cb_c..cb_o, theta_c..theta_o, wc_f,wc_i,wc_o},
    // g2{...}, lin_b, lin_w, x
    p.c1b = F(2);  p.c1w = F(3);  p.c2b = F(4);  p.c2w = F(5);
    p.Wc1 = F(6);  p.Wi1 = F(8);  p.Wo1 = F(9);
    p.bc1 = F(10); p.bi1 = F(12); p.bo1 = F(13);
    p.cbc1 = F(14); p.cbi1 = F(16); p.cbo1 = F(17);
    p.wco1 = F(24);
    p.Wc2 = F(25); p.Wi2 = F(27); p.Wo2 = F(28);
    p.bc2 = F(29); p.bi2 = F(31); p.bo2 = F(32);
    p.cbc2 = F(33); p.cbi2 = F(35); p.cbo2 = F(36);
    p.wco2 = F(43);
    p.linb = F(44); p.linw = F(45); p.x = F(46);
  } else {
    // insertion: g1{W_i,theta_i,cb_i,b_i, W_f..., W_c..., W_o..., wc_i,wc_f,wc_o},
    // g2{...}, c1_w, c1_b, c2_w, c2_b, lin_w, lin_b, x, edge_index, edge_weight
    p.Wi1 = F(0);  p.cbi1 = F(2);  p.bi1 = F(3);
    p.Wc1 = F(8);  p.cbc1 = F(10); p.bc1 = F(11);
    p.Wo1 = F(12); p.cbo1 = F(14); p.bo1 = F(15);
    p.wco1 = F(18);
    p.Wi2 = F(19); p.cbi2 = F(21); p.bi2 = F(22);
    p.Wc2 = F(27); p.cbc2 = F(29); p.bc2 = F(30);
    p.Wo2 = F(31); p.cbo2 = F(33); p.bo2 = F(34);
    p.wco2 = F(37);
    p.c1w = F(38); p.c1b = F(39); p.c2w = F(40); p.c2b = F(41);
    p.linw = F(42); p.linb = F(43); p.x = F(44);
  }
  int blocks = (TILES + WAVES_PER_BLOCK - 1) / WAVES_PER_BLOCK;  // 782
  gclstm_fused_kernel<<<blocks, BLOCK_THREADS, 0, stream>>>(p, (float*)d_out);
}
